// Quantizer_16784732192795
// MI455X (gfx1250) — matrix operand order repaired and checked by reading, not compiled, not run on hardware
//
#include <hip/hip_runtime.h>

// ---------------------------------------------------------------------------
// CDNA5 (gfx1250) VQ-VAE quantizer.
//   argmin_k ||z - e_k||^2  ==  argmin_k ( ||e_k||^2 - 2 z.e_k )   (||z||^2 const/row)
// The whole biased distance d2' = -2 z.e + ||e||^2 + 0.125 is produced directly
// by bf16 WMMA over an augmented K=96 problem:
//   A dims 0..63 = z (bf16), dim64 = dim65 = 1.0, rest 0
//   B dims 0..63 = -2*e (bf16), dim64/65 = hi/lo bf16 split of ||e||^2+0.125
// d2' > 0, so its f32 bits order like u32: argmin is tracked with
//   packed = (bits(d2') & ~2047) | code   ->  v_and + v_or3 + v_min3_u32
// Tiles are processed in pairs with independent accumulators so the second
// tile's WMMAs fill the first tile's WMMA->VALU hazard slots (no v_nop).
// z_q is gathered from the fp32 codebook, so output values are exact fp32.
// ---------------------------------------------------------------------------

typedef __attribute__((ext_vector_type(16))) __bf16 v16bf;
typedef __attribute__((ext_vector_type(8)))  float  v8f;

union BV {
  uint4 q[2];
  unsigned u32v[8];
  unsigned short us[16];
  v16bf bf;
};

__device__ __forceinline__ unsigned short f2bf(float f) {
  unsigned u = __float_as_uint(f);
  u += 0x7FFFu + ((u >> 16) & 1u);   // round-to-nearest-even
  return (unsigned short)(u >> 16);
}
__device__ __forceinline__ float bf2f(unsigned short h) {
  return __uint_as_float(((unsigned)h) << 16);
}

#define D_MODEL        64
#define WAVES_PER_WG   8
#define ROWS_PER_WG    128        // 8 waves * 16 rows
#define BIAS           0.125f
#define LANE_B_G       80         // bytes/lane/tile in Bbuf: 2x32B (-2e) + 16B (e2 hi/lo)
#define TILE_B_G       2560       // 32 lanes * 80B
#define LANE_B_LDS     96         // padded LDS lane stride
#define TILE_B_LDS     3072       // 32 * 96
#define CHUNK_TILES    8          // 128 codes staged per chunk
#define CHUNK_B_G      (CHUNK_TILES * TILE_B_G)     // 20480

// --- Preprocess: pack codebook into WMMA B-operand layout (bf16) ------------
// B fragment (32x16 16-bit): lane n<16 holds column n, elems = K 0..7,16..23 of the
// 32-chunk; lane 16+n holds K 8..15,24..31. Per tile, per lane, 40 bf16:
//   j  0..15 -> WMMA#1 (dims 0..31):  -2*e
//   j 16..31 -> WMMA#2 (dims 32..63): -2*e
//   j 32..39 -> WMMA#3 first dwords: half==0: elem0=hi(e2+bias), elem1=lo, rest 0
__global__ void prep_codebook(const float* __restrict__ emb,
                              unsigned short* __restrict__ Bbuf, int K) {
  int e = blockIdx.x * blockDim.x + threadIdx.x;
  int total = (K / 16) * 1280;                        // 32 lanes * 40 ush
  if (e >= total) return;
  int tile = e / 1280;
  int r    = e % 1280;
  int lane = r / 40;
  int j    = r % 40;
  int code = tile * 16 + (lane & 15);
  int half = lane >> 4;
  unsigned short val = 0;
  if (j < 16) {
    int k = (j < 8) ? half * 8 + j : 16 + half * 8 + (j - 8);
    val = f2bf(-2.0f * emb[code * D_MODEL + k]);
  } else if (j < 32) {
    int k = 32 + ((j < 24) ? half * 8 + (j - 16) : 16 + half * 8 + (j - 24));
    val = f2bf(-2.0f * emb[code * D_MODEL + k]);
  } else if (half == 0) {
    int kp = j - 32;                 // dims 64..71; only 64/65 nonzero
    if (kp == 0 || kp == 1) {
      float s = 0.f;
      #pragma unroll
      for (int k = 0; k < D_MODEL; ++k) {
        float v = emb[code * D_MODEL + k]; s += v * v;
      }
      float t = s + BIAS;
      unsigned short hi = f2bf(t);
      val = (kp == 0) ? hi : f2bf(t - bf2f(hi));   // hi/lo split keeps e2 exact-ish
    }
  }
  Bbuf[e] = val;
}

__global__ void zero_loss(float* p) { *p = 0.f; }

// --- Main kernel ------------------------------------------------------------
__global__ __launch_bounds__(256) void vq_kernel(
    const float* __restrict__ z_e,
    const float* __restrict__ emb,
    const unsigned short* __restrict__ Bbuf,
    float* __restrict__ out_zq,
    float* __restrict__ out_loss,
    float* __restrict__ out_idx,
    int K, float lossScale)
{
  __shared__ __align__(16) unsigned char sB[CHUNK_TILES * TILE_B_LDS];  // 24KB
  __shared__ int sIdx[WAVES_PER_WG][16];

  const int tid  = threadIdx.x;
  const int wave = tid >> 5;
  const int lane = tid & 31;
  const int half = lane >> 4;
  const int col  = lane & 15;
  const int wrow = (blockIdx.x * WAVES_PER_WG + wave) * 16;  // 16-row A tile

  // ---- A fragments (fp32 -> bf16, ISA A layout) ----
  // lane<16: elems0..7=K0..7, elems8..15=K16..23 ; lane>=16: K8..15 / K24..31
  BV a1, a2, a3;
  {
    const float* zr = z_e + (size_t)(wrow + col) * D_MODEL + half * 8;
    #pragma unroll
    for (int c = 0; c < 4; ++c) {
      float4 f0 = *(const float4*)(zr + c * 16);
      float4 f1 = *(const float4*)(zr + c * 16 + 4);
      unsigned short* dst = (c < 2 ? a1.us : a2.us) + (c & 1) * 8;
      dst[0] = f2bf(f0.x); dst[1] = f2bf(f0.y); dst[2] = f2bf(f0.z); dst[3] = f2bf(f0.w);
      dst[4] = f2bf(f1.x); dst[5] = f2bf(f1.y); dst[6] = f2bf(f1.z); dst[7] = f2bf(f1.w);
    }
    #pragma unroll
    for (int i = 0; i < 8; ++i) a3.u32v[i] = 0u;
    if (half == 0) { a3.us[0] = 0x3F80; a3.us[1] = 0x3F80; }  // dims 64,65 = 1.0
  }

  BV b3a, b3b;                 // only dword0 varies (e2 hi/lo); rest constant 0
  #pragma unroll
  for (int i = 0; i < 8; ++i) { b3a.u32v[i] = 0u; b3b.u32v[i] = 0u; }

  unsigned pmin[8];
  #pragma unroll
  for (int r = 0; r < 8; ++r) pmin[r] = 0xFFFFFFFFu;

  const int nChunks = K / (CHUNK_TILES * 16);
  for (int chunk = 0; chunk < nChunks; ++chunk) {
    __syncthreads();
    // Async-stage 20KB of B (128 codes): 256 threads x 80B, global -> LDS
    // directly via the CDNA5 async-to-LDS path (ASYNCcnt-tracked).
    {
      unsigned ldsOff = (unsigned)(size_t)(sB + (tid >> 5) * TILE_B_LDS +
                                                (tid & 31) * LANE_B_LDS);
      unsigned vOff = (unsigned)(tid * LANE_B_G);
      unsigned long long sbase =
          (unsigned long long)(size_t)((const char*)Bbuf + (size_t)chunk * CHUNK_B_G);
      asm volatile(
        "global_load_async_to_lds_b128 %0, %1, %2 offset:0\n\t"
        "global_load_async_to_lds_b128 %0, %1, %2 offset:16\n\t"
        "global_load_async_to_lds_b128 %0, %1, %2 offset:32\n\t"
        "global_load_async_to_lds_b128 %0, %1, %2 offset:48\n\t"
        "global_load_async_to_lds_b128 %0, %1, %2 offset:64"
        :: "v"(ldsOff), "v"(vOff), "s"(sbase) : "memory");
      if (chunk + 1 < nChunks)   // global_prefetch_b8 of next chunk
        __builtin_prefetch((const char*)Bbuf + (size_t)(chunk + 1) * CHUNK_B_G
                           + tid * LANE_B_G, 0, 1);
      asm volatile("s_wait_asynccnt 0" ::: "memory");
    }
    __syncthreads();

    #pragma unroll
    for (int tp = 0; tp < CHUNK_TILES; tp += 2) {
      const unsigned char* pa = sB + (tp + 0) * TILE_B_LDS + lane * LANE_B_LDS;
      const unsigned char* pb = sB + (tp + 1) * TILE_B_LDS + lane * LANE_B_LDS;
      BV b1a, b2a, b1b, b2b;
      b1a.q[0] = ((const uint4*)pa)[0]; b1a.q[1] = ((const uint4*)pa)[1];
      b2a.q[0] = ((const uint4*)pa)[2]; b2a.q[1] = ((const uint4*)pa)[3];
      b3a.u32v[0] = *(const unsigned*)(pa + 64);        // ds_load_b32
      b1b.q[0] = ((const uint4*)pb)[0]; b1b.q[1] = ((const uint4*)pb)[1];
      b2b.q[0] = ((const uint4*)pb)[2]; b2b.q[1] = ((const uint4*)pb)[3];
      b3b.u32v[0] = *(const unsigned*)(pb + 64);

      v8f ca = {}, cb = {};
      ca = __builtin_amdgcn_wmma_f32_16x16x32_bf16(false, a1.bf, false, b1a.bf,
                                                   (short)0, ca, false, false);
      ca = __builtin_amdgcn_wmma_f32_16x16x32_bf16(false, a2.bf, false, b2a.bf,
                                                   (short)0, ca, false, false);
      ca = __builtin_amdgcn_wmma_f32_16x16x32_bf16(false, a3.bf, false, b3a.bf,
                                                   (short)0, ca, false, false);
      cb = __builtin_amdgcn_wmma_f32_16x16x32_bf16(false, a1.bf, false, b1b.bf,
                                                   (short)0, cb, false, false);
      cb = __builtin_amdgcn_wmma_f32_16x16x32_bf16(false, a2.bf, false, b2b.bf,
                                                   (short)0, cb, false, false);
      cb = __builtin_amdgcn_wmma_f32_16x16x32_bf16(false, a3.bf, false, b3b.bf,
                                                   (short)0, cb, false, false);

      // c[r] = ||e||^2 + 0.125 - 2 z.e  (> 0)  for row (lane>>4)*8+r, code col
      unsigned codeA = (unsigned)(chunk * (CHUNK_TILES * 16) + (tp + 0) * 16 + col);
      unsigned codeB = codeA + 16u;
      #pragma unroll
      for (int r = 0; r < 8; ++r) {
        unsigned pA = (__float_as_uint(ca[r]) & 0xFFFFF800u) | codeA;
        unsigned pB = (__float_as_uint(cb[r]) & 0xFFFFF800u) | codeB;
        unsigned t  = pA < pB ? pA : pB;
        pmin[r] = t < pmin[r] ? t : pmin[r];        // -> v_min3_u32
      }
    }
  }

  // Min-reduce packed (value|index) over the 16 code columns of each half.
  #pragma unroll
  for (int m = 1; m < 16; m <<= 1) {
    #pragma unroll
    for (int r = 0; r < 8; ++r) {
      unsigned o = (unsigned)__shfl_xor((int)pmin[r], m, 32);
      pmin[r] = o < pmin[r] ? o : pmin[r];
    }
  }
  if (lane == 0) {
    #pragma unroll
    for (int r = 0; r < 8; ++r) sIdx[wave][r] = (int)(pmin[r] & 2047u);
  }
  if (lane == 16) {
    #pragma unroll
    for (int r = 0; r < 8; ++r) sIdx[wave][8 + r] = (int)(pmin[r] & 2047u);
  }
  __syncthreads();

  // Gather fp32 codebook rows (z_q == z_q_st numerically), loss, indices.
  {
    int r  = lane >> 1;
    int d0 = (lane & 1) * 32;
    int q  = sIdx[wave][r];
    size_t row = (size_t)(wrow + r);
    const float4* ep = (const float4*)(emb    + (size_t)q * D_MODEL + d0);
    const float4* zp = (const float4*)(z_e    + row * D_MODEL + d0);
    float4*       op = (float4*)      (out_zq + row * D_MODEL + d0);
    float lacc = 0.f;
    #pragma unroll
    for (int i = 0; i < 8; ++i) {
      float4 e = ep[i], z = zp[i];
      op[i] = e;
      float dx = e.x - z.x, dy = e.y - z.y, dz = e.z - z.z, dw = e.w - z.w;
      lacc += dx * dx + dy * dy + dz * dz + dw * dw;
    }
    if (lane < 16) out_idx[wrow + lane] = (float)sIdx[wave][lane];
    #pragma unroll
    for (int m = 1; m < 32; m <<= 1) lacc += __shfl_xor(lacc, m, 32);
    if (lane == 0) atomicAdd(out_loss, lacc * lossScale);
  }
}

// ---------------------------------------------------------------------------
extern "C" void kernel_launch(void* const* d_in, const int* in_sizes, int n_in,
                              void* d_out, int out_size, void* d_ws, size_t ws_size,
                              hipStream_t stream) {
  const float* z_e = (const float*)d_in[0];
  const float* emb = (const float*)d_in[1];
  const int nZ    = in_sizes[0];            // 4*32768*64
  const int K     = in_sizes[1] / D_MODEL;  // 2048
  const int nRows = nZ / D_MODEL;           // 131072

  unsigned short* Bbuf = (unsigned short*)d_ws;   // (K/16)*2560 bytes = 320KB

  float* out      = (float*)d_out;
  float* out_zq   = out;                    // [nRows*64]
  float* out_loss = out + nZ;               // [1]
  float* out_idx  = out + nZ + 1;           // [nRows]

  const int totalB = (K / 16) * 1280;       // ushorts in Bbuf
  prep_codebook<<<(totalB + 255) / 256, 256, 0, stream>>>(emb, Bbuf, K);
  zero_loss<<<1, 1, 0, stream>>>(out_loss);

  const float lossScale = 1.25f / (float)nZ;  // (1+BETA) * mean
  vq_kernel<<<nRows / ROWS_PER_WG, 256, 0, stream>>>(
      z_e, emb, Bbuf, out_zq, out_loss, out_idx, K, lossScale);
}
